// PoetryModel_90357521973892
// MI455X (gfx1250) — matrix operand order repaired and checked by reading, not compiled, not run on hardware
//
#include <hip/hip_runtime.h>
#include <math.h>

typedef unsigned short ushort_t;
typedef __attribute__((ext_vector_type(16))) __bf16 v16bf;
typedef __attribute__((ext_vector_type(8)))  float  v8f;
typedef __attribute__((ext_vector_type(4)))  int    v4i;

union Frag { v16bf v; uint4 q[2]; };

#define WMMA_BF16(a, b, c) \
  __builtin_amdgcn_wmma_f32_16x16x32_bf16(false, (a), false, (b), (short)0, (c), false, false)

#if __has_builtin(__builtin_amdgcn_global_load_async_to_lds_b128) && \
    __has_builtin(__builtin_amdgcn_s_wait_asynccnt)
#define HAVE_ASYNC_LDS 1
#endif

#if __has_builtin(__builtin_amdgcn_sched_barrier)
#define SCHED_FENCE() __builtin_amdgcn_sched_barrier(0)
#else
#define SCHED_FENCE()
#endif

// 16-byte global -> LDS copy; CDNA5 async-DMA path when available.
__device__ __forceinline__ void copy16_g2l(ushort_t* lds, const ushort_t* gsrc) {
#ifdef HAVE_ASYNC_LDS
  __builtin_amdgcn_global_load_async_to_lds_b128(
      (__attribute__((address_space(1))) v4i*)(uintptr_t)gsrc,
      (__attribute__((address_space(3))) v4i*)(unsigned)(uintptr_t)lds,
      0, 0);
#else
  *(uint4*)lds = *(const uint4*)gsrc;
#endif
}
__device__ __forceinline__ void stage_wait() {
#ifdef HAVE_ASYNC_LDS
  __builtin_amdgcn_s_wait_asynccnt(0);
#endif
}

__device__ __forceinline__ ushort_t f2bf(float f) {
  unsigned u = __float_as_uint(f);
  unsigned r = u + 0x7FFFu + ((u >> 16) & 1u);   // round-to-nearest-even
  return (ushort_t)(r >> 16);
}

// A fragment: 16x32 bf16, row-major, lda in elements.
__device__ __forceinline__ v16bf load_a_frag(const ushort_t* A, int lda) {
  int lane = threadIdx.x & 31;
  int m = lane & 15, hi = lane >> 4;
  const ushort_t* p = A + (size_t)m * lda + hi * 8;
  Frag f;
  f.q[0] = *(const uint4*)(p);
  f.q[1] = *(const uint4*)(p + 16);
  return f.v;
}

// B fragment: 32x16 bf16 from pre-transposed B^T [N,K] row-major, ldb in elements.
__device__ __forceinline__ v16bf load_b_frag(const ushort_t* BT, int ldb) {
  int lane = threadIdx.x & 31;
  int n = lane & 15, hi = lane >> 4;
  const ushort_t* p = BT + (size_t)n * ldb + hi * 8;
  Frag f;
  f.q[0] = *(const uint4*)(p);
  f.q[1] = *(const uint4*)(p + 16);
  return f.v;
}

// ---------------------------------------------------------------- prep kernels

__global__ void k_init_bar(int* bar) { *bar = 0; }

__global__ void k_gather_embed(const int* __restrict__ X,
                               const float* __restrict__ embed,
                               ushort_t* __restrict__ Xe) {
  int row = blockIdx.x;               // 0..8191  (s*64 + b)
  int tok = X[row];
  Xe[(size_t)row * 256 + threadIdx.x] = f2bf(embed[(size_t)tok * 256 + threadIdx.x]);
}

// dst[n*R + r] = bf16(src[r*C + n])
__global__ void k_transpose_bf16(const float* __restrict__ src,
                                 ushort_t* __restrict__ dst, int R, int C) {
  int n = blockIdx.x;
  for (int r = threadIdx.x; r < R; r += blockDim.x)
    dst[(size_t)n * R + r] = f2bf(src[(size_t)r * C + n]);
}

// ------------------------------------------------------------- LSTM recurrence
// Persistent grid: 64 WGs x 256 threads. WG owns 16 H-columns x 32 batch rows,
// all 4 gates. Per step: async-stage x_t/h_t into LDS, WMMA with t-invariant
// register-resident B fragments; 3 rotating A-fragment groups + sched barriers
// keep 16-24 DS ops in flight so the waitcnt pass emits partial s_wait_dscnt.

#define RWG 64
#define PADX 264   // 256 + 8 ushorts
#define PADH 520   // 512 + 8

__global__ __launch_bounds__(256) void k_lstm_recurrence(
    const ushort_t* __restrict__ Xe,    // [128*64, 256]
    const ushort_t* __restrict__ WxT,   // [4][512(n)][256(k)]
    const ushort_t* __restrict__ WhT,   // [4][512(n)][512(k)]
    const float* __restrict__ b_i, const float* __restrict__ b_f,
    const float* __restrict__ b_c, const float* __restrict__ b_o,
    ushort_t* __restrict__ hping,       // [2][64][512] bf16
    ushort_t* __restrict__ Hs,          // [128][64][512] bf16
    float* __restrict__ ht_out, float* __restrict__ ct_out,
    int* __restrict__ bar) {
  const int wg = blockIdx.x, tid = threadIdx.x;
  const int n0 = (wg & 31) * 16;        // H-column tile
  const int r0 = (wg >> 5) * 32;        // batch rows [r0, r0+32)
  const int wave = tid >> 5;
  const int g = wave >> 1;              // gate for this wave
  const int rtl = wave & 1;             // local row tile (16 rows)

  __shared__ ushort_t sX[32 * PADX];
  __shared__ ushort_t sH[32 * PADH];
  __shared__ float gl[4][32][16];

  // t-invariant B fragments, register-resident (192 VGPRs/wave)
  v16bf Bx[8], Bh[16];
  {
    const ushort_t* BxP = WxT + ((size_t)g * 512 + n0) * 256;
#pragma unroll
    for (int kk = 0; kk < 8; kk++) Bx[kk] = load_b_frag(BxP + kk * 32, 256);
    const ushort_t* BhP = WhT + ((size_t)g * 512 + n0) * 512;
#pragma unroll
    for (int kk = 0; kk < 16; kk++) Bh[kk] = load_b_frag(BhP + kk * 32, 512);
  }

  const float* bp[4] = {b_i, b_f, b_c, b_o};
  float c_reg[2], bias[4][2];
  int em[2], ej[2];
#pragma unroll
  for (int e = 0; e < 2; e++) {
    int idx = tid * 2 + e;              // 0..511 over [32 rows x 16 cols]
    em[e] = idx >> 4; ej[e] = idx & 15; c_reg[e] = 0.f;
#pragma unroll
    for (int gg = 0; gg < 4; gg++) bias[gg][e] = bp[gg][n0 + ej[e]];
  }
#pragma unroll
  for (int e = 0; e < 2; e++) hping[(size_t)(r0 + em[e]) * 512 + n0 + ej[e]] = 0;

  __syncthreads();
  if (tid == 0) {
    __hip_atomic_fetch_add(bar, 1, __ATOMIC_ACQ_REL, __HIP_MEMORY_SCOPE_AGENT);
    while (__hip_atomic_load(bar, __ATOMIC_ACQUIRE, __HIP_MEMORY_SCOPE_AGENT) < RWG)
      __builtin_amdgcn_s_sleep(2);
  }
  __syncthreads();
  int epoch = 1;

  for (int t = 0; t < 128; t++) {
    const ushort_t* hsrc = hping + (size_t)(t & 1) * (64 * 512);
    const ushort_t* xsrc = Xe + (size_t)(t * 64 + r0) * 256;

    // ---- bulk async-stage x_t and h_t rows [r0,r0+32) into LDS
#pragma unroll
    for (int i = 0; i < 4; i++) {       // 1024 x 16B chunks of x
      int idx = tid + i * 256;
      int r = idx >> 5, cc = (idx & 31) * 8;
      copy16_g2l(&sX[r * PADX + cc], xsrc + (size_t)r * 256 + cc);
    }
#pragma unroll
    for (int i = 0; i < 8; i++) {       // 2048 x 16B chunks of h
      int idx = tid + i * 256;
      int r = idx >> 6, cc = (idx & 63) * 8;
      copy16_g2l(&sH[r * PADH + cc], hsrc + (size_t)(r0 + r) * 512 + cc);
    }
    stage_wait();
    __syncthreads();

    // ---- GEMM: one 16x16 tile per wave (gate g, rows r0+rtl*16), K=256+512
    {
      v8f acc = {0.f, 0.f, 0.f, 0.f, 0.f, 0.f, 0.f, 0.f};
      const ushort_t* aX = sX + rtl * 16 * PADX;
      const ushort_t* aH = sH + rtl * 16 * PADH;
      v16bf f0[4], f1[4], f2[4];
      // prime three groups (24 ds_load_b128 in flight)
#pragma unroll
      for (int i = 0; i < 4; i++) f0[i] = load_a_frag(aX + i * 32, PADX);
#pragma unroll
      for (int i = 0; i < 4; i++) f1[i] = load_a_frag(aX + (4 + i) * 32, PADX);
#pragma unroll
      for (int i = 0; i < 4; i++) f2[i] = load_a_frag(aH + i * 32, PADH);
      SCHED_FENCE();
#pragma unroll
      for (int i = 0; i < 4; i++) acc = WMMA_BF16(f0[i], Bx[i], acc);
      SCHED_FENCE();
#pragma unroll
      for (int i = 0; i < 4; i++) f0[i] = load_a_frag(aH + (4 + i) * 32, PADH);
      SCHED_FENCE();
#pragma unroll
      for (int i = 0; i < 4; i++) acc = WMMA_BF16(f1[i], Bx[4 + i], acc);
      SCHED_FENCE();
#pragma unroll
      for (int i = 0; i < 4; i++) f1[i] = load_a_frag(aH + (8 + i) * 32, PADH);
      SCHED_FENCE();
#pragma unroll
      for (int i = 0; i < 4; i++) acc = WMMA_BF16(f2[i], Bh[i], acc);
      SCHED_FENCE();
#pragma unroll
      for (int i = 0; i < 4; i++) f2[i] = load_a_frag(aH + (12 + i) * 32, PADH);
      SCHED_FENCE();
#pragma unroll
      for (int i = 0; i < 4; i++) acc = WMMA_BF16(f0[i], Bh[4 + i], acc);
      SCHED_FENCE();
#pragma unroll
      for (int i = 0; i < 4; i++) acc = WMMA_BF16(f1[i], Bh[8 + i], acc);
      SCHED_FENCE();
#pragma unroll
      for (int i = 0; i < 4; i++) acc = WMMA_BF16(f2[i], Bh[12 + i], acc);

      int lane = tid & 31, n = lane & 15, mb = (lane >> 4) * 8;
#pragma unroll
      for (int r = 0; r < 8; r++) gl[g][rtl * 16 + mb + r][n] = acc[r];
    }
    __syncthreads();

    // ---- cell update (c stays in registers)
#pragma unroll
    for (int e = 0; e < 2; e++) {
      int m = em[e], j = ej[e], col = n0 + j, row = r0 + m;
      float gi = gl[0][m][j] + bias[0][e];
      float gf = gl[1][m][j] + bias[1][e];
      float gc = gl[2][m][j] + bias[2][e];
      float go = gl[3][m][j] + bias[3][e];
      float it = 1.f / (1.f + __expf(-gi));
      float ft = 1.f / (1.f + __expf(-gf));
      float cc = tanhf(gc);
      float ot = 1.f / (1.f + __expf(-go));
      float c  = ft * c_reg[e] + it * cc;
      c_reg[e] = c;
      float h  = ot * tanhf(c);
      ushort_t hb = f2bf(h);
      hping[(size_t)((t + 1) & 1) * (64 * 512) + (size_t)row * 512 + col] = hb;
      Hs[(size_t)t * (64 * 512) + (size_t)row * 512 + col] = hb;
      if (t == 127) { ht_out[row * 512 + col] = h; ct_out[row * 512 + col] = c; }
    }

    // ---- device-scope barrier before next step reads full h
    __syncthreads();
    if (tid == 0) {
      __hip_atomic_fetch_add(bar, 1, __ATOMIC_ACQ_REL, __HIP_MEMORY_SCOPE_AGENT);
      int target = RWG * (epoch + 1);
      while (__hip_atomic_load(bar, __ATOMIC_ACQUIRE, __HIP_MEMORY_SCOPE_AGENT) < target)
        __builtin_amdgcn_s_sleep(2);
    }
    __syncthreads();
    epoch++;
  }
}

// -------------------------------------------------- output projection (68.7 GF)
// out[8192,8192] = Hs[8192,512](bf16) @ WT[8192,512](bf16) + b.
// WG tile 128x128, wave tile 32x64. Async double-buffered LDS staging of
// 128x32 A/B slices; all 6 fragments loaded before the 8 WMMAs of a k-step.

#define SPAD 40   // 32 + 8 ushorts per staged row

union OutSmem {
  ushort_t stage[2][2][128 * SPAD];   // [buf][0=A,1=B]  -> 40 KB
  float st[128][128];                 // 64 KB epilogue
};

__global__ __launch_bounds__(256) void k_out_gemm(
    const ushort_t* __restrict__ Hs, const ushort_t* __restrict__ WT,
    const float* __restrict__ bias, float* __restrict__ out) {
  __shared__ OutSmem smem;
  const int tid = threadIdx.x;
  const int row0 = blockIdx.y * 128, col0 = blockIdx.x * 128;
  const int wave = tid >> 5;
  const int wm = wave >> 1, wn = wave & 1;

  v8f acc[2][4];
#pragma unroll
  for (int i = 0; i < 2; i++)
#pragma unroll
    for (int j = 0; j < 4; j++)
      acc[i][j] = (v8f){0.f, 0.f, 0.f, 0.f, 0.f, 0.f, 0.f, 0.f};

  auto do_stage = [&](int buf, int k) {
#pragma unroll
    for (int i = 0; i < 2; i++) {
      int idx = tid * 2 + i;            // 512 x 16B chunks per matrix
      int r = idx >> 2, cc = (idx & 3) * 8;
      copy16_g2l(&smem.stage[buf][0][r * SPAD + cc],
                 Hs + (size_t)(row0 + r) * 512 + k + cc);
      copy16_g2l(&smem.stage[buf][1][r * SPAD + cc],
                 WT + (size_t)(col0 + r) * 512 + k + cc);
    }
  };

  do_stage(0, 0);
  stage_wait();
  __syncthreads();

  for (int ks = 0; ks < 16; ks++) {
    int cur = ks & 1;
    if (ks < 15) do_stage(cur ^ 1, (ks + 1) * 32);   // DMA next slice underneath

    const ushort_t* sA = smem.stage[cur][0];
    const ushort_t* sB = smem.stage[cur][1];
    v16bf a0 = load_a_frag(sA + (wm * 32) * SPAD, SPAD);
    v16bf a1 = load_a_frag(sA + (wm * 32 + 16) * SPAD, SPAD);
    v16bf bf[4];
#pragma unroll
    for (int j = 0; j < 4; j++)
      bf[j] = load_b_frag(sB + (wn * 64 + j * 16) * SPAD, SPAD);
    SCHED_FENCE();
#pragma unroll
    for (int j = 0; j < 4; j++) {
      acc[0][j] = WMMA_BF16(a0, bf[j], acc[0][j]);
      acc[1][j] = WMMA_BF16(a1, bf[j], acc[1][j]);
    }
    stage_wait();
    __syncthreads();
  }

  // epilogue: stage C through LDS, write coalesced b128 stores with bias
  int lane = tid & 31, n = lane & 15, mb = (lane >> 4) * 8;
#pragma unroll
  for (int i = 0; i < 2; i++)
#pragma unroll
    for (int j = 0; j < 4; j++)
#pragma unroll
      for (int r = 0; r < 8; r++)
        smem.st[wm * 32 + i * 16 + mb + r][wn * 64 + j * 16 + n] = acc[i][j][r];
  __syncthreads();

#pragma unroll
  for (int pass = 0; pass < 16; pass++) {
    int row = pass * 8 + (tid >> 5);
    int qc = (tid & 31) * 4;
    float4 v = *(const float4*)&smem.st[row][qc];
    int col = col0 + qc;
    v.x += bias[col]; v.y += bias[col + 1]; v.z += bias[col + 2]; v.w += bias[col + 3];
    *(float4*)&out[(size_t)(row0 + row) * 8192 + col] = v;
  }
}

// --------------------------------------------------------------------- launch

extern "C" void kernel_launch(void* const* d_in, const int* in_sizes, int n_in,
                              void* d_out, int out_size, void* d_ws, size_t ws_size,
                              hipStream_t stream) {
  (void)in_sizes; (void)n_in; (void)out_size; (void)ws_size;
  const int*   X     = (const int*)d_in[0];
  const float* embed = (const float*)d_in[1];
  const float* W     = (const float*)d_in[2];
  const float* b     = (const float*)d_in[3];
  const float* Wx[4] = {(const float*)d_in[4],  (const float*)d_in[7],
                        (const float*)d_in[10], (const float*)d_in[13]};
  const float* Wh[4] = {(const float*)d_in[5],  (const float*)d_in[8],
                        (const float*)d_in[11], (const float*)d_in[14]};
  const float* bg[4] = {(const float*)d_in[6],  (const float*)d_in[9],
                        (const float*)d_in[12], (const float*)d_in[15]};

  char* ws = (char*)d_ws;
  size_t off = 0;
  ushort_t* Xe  = (ushort_t*)(ws + off); off += (size_t)8192 * 256 * 2;
  ushort_t* WxT = (ushort_t*)(ws + off); off += (size_t)4 * 512 * 256 * 2;
  ushort_t* WhT = (ushort_t*)(ws + off); off += (size_t)4 * 512 * 512 * 2;
  ushort_t* WT  = (ushort_t*)(ws + off); off += (size_t)8192 * 512 * 2;
  ushort_t* hp  = (ushort_t*)(ws + off); off += (size_t)2 * 64 * 512 * 2;
  ushort_t* Hs  = (ushort_t*)(ws + off); off += (size_t)128 * 64 * 512 * 2;
  int*      bar = (int*)(ws + off);

  float* out    = (float*)d_out;
  float* ht_out = out + (size_t)8192 * 8192;
  float* ct_out = ht_out + 64 * 512;

  k_init_bar<<<1, 1, 0, stream>>>(bar);
  k_gather_embed<<<8192, 256, 0, stream>>>(X, embed, Xe);
  for (int g = 0; g < 4; g++) {
    k_transpose_bf16<<<512, 256, 0, stream>>>(Wx[g], WxT + (size_t)g * 512 * 256, 256, 512);
    k_transpose_bf16<<<512, 256, 0, stream>>>(Wh[g], WhT + (size_t)g * 512 * 512, 512, 512);
  }
  k_transpose_bf16<<<8192, 256, 0, stream>>>(W, WT, 512, 8192);

  k_lstm_recurrence<<<RWG, 256, 0, stream>>>(Xe, WxT, WhT,
                                             bg[0], bg[1], bg[2], bg[3],
                                             hp, Hs, ht_out, ct_out, bar);
  k_out_gemm<<<dim3(64, 64), 256, 0, stream>>>(Hs, WT, b, out);
}
